// MultiHeadAttention_21827023798568
// MI455X (gfx1250) — matrix operand order repaired and checked by reading, not compiled, not run on hardware
//
#include <hip/hip_runtime.h>

// ---------------- problem constants ----------------
#define N_HEADS   16
#define HEAD_DIM  64
#define D_MODEL   1024
#define BATCH     2
#define SEQ       2048
#define ROWS      (BATCH * SEQ)          // 4096 total query/key rows

typedef __attribute__((ext_vector_type(16))) __bf16 v16bf;
typedef __attribute__((ext_vector_type(8)))  float  v8f;

// ---------------- helpers ----------------
static __device__ __forceinline__ unsigned short f2bf(float f) {
  union { float f; unsigned u; } v; v.f = f;
  unsigned r = v.u + 0x7FFFu + ((v.u >> 16) & 1u);   // round-to-nearest-even
  return (unsigned short)(r >> 16);
}

union FragU { uint4 q[2]; v16bf v; };

// Load a 16-bf16-per-lane WMMA operand fragment (tile pre-swizzled in memory:
// lane's 16 elements are contiguous -> two coalesced b128 loads).
static __device__ __forceinline__ v16bf load_frag(const unsigned short* tileBase, int lane) {
  const uint4* p = reinterpret_cast<const uint4*>(tileBase) + lane * 2;
  FragU u; u.q[0] = p[0]; u.q[1] = p[1];
  return u.v;
}

static __device__ __forceinline__ v8f wmma_bf16(v16bf a, v16bf b, v8f c) {
  return __builtin_amdgcn_wmma_f32_16x16x32_bf16(false, a, false, b, (short)0, c, false, false);
}

// Element offset inside A-fragment-swizzled storage of an MxK bf16 matrix.
// Tiles are 16x32, row-major tile order; within a tile this code's contiguous-k layout:
//   lanes 0-15: K 0-15; lanes 16-31: K 16-31; VGPR j holds K pair (2j,2j+1).
static __device__ __forceinline__ size_t offA(int m, int k, int K) {
  int tile = (m >> 4) * (K >> 5) + (k >> 5);
  int mi = m & 15, ki = k & 31;
  int sel = (ki >> 4) & 1;
  int grp = ki >> 4;
  int v = (ki & 15) >> 1;
  int lane = mi + 16 * sel;
  return (size_t)tile * 512 + lane * 16 + v * 2 + (ki & 1);
}

// Element offset inside B-fragment-swizzled storage of a KxN bf16 matrix.
// Tiles are 32x16, tile = nt*(K/32)+kt; contiguous-k layout (same as A):
//   lanes 0-15: K 0-15; lanes 16-31: K 16-31; VGPR j holds K pair (2j,2j+1).
static __device__ __forceinline__ size_t offB(int k, int n, int K) {
  int tile = (n >> 4) * (K >> 5) + (k >> 5);
  int ki = k & 31;
  int lane = (n & 15) + 16 * (ki >> 4);
  int v = (ki & 15) >> 1;
  return (size_t)tile * 512 + lane * 16 + v * 2 + (ki & 1);
}

static __device__ __forceinline__ float redmax16(float v) {
  v = fmaxf(v, __shfl_xor(v, 1, 32));
  v = fmaxf(v, __shfl_xor(v, 2, 32));
  v = fmaxf(v, __shfl_xor(v, 4, 32));
  v = fmaxf(v, __shfl_xor(v, 8, 32));
  return v;
}
static __device__ __forceinline__ float redsum16(float v) {
  v += __shfl_xor(v, 1, 32);
  v += __shfl_xor(v, 2, 32);
  v += __shfl_xor(v, 4, 32);
  v += __shfl_xor(v, 8, 32);
  return v;
}

// ---------------- stage 1: fp32 -> bf16 fragment packers ----------------
// One thread per OUTPUT element (coalesced writes, gathered reads hit L2).
__global__ void pack_a_kernel(const float* __restrict__ in,
                              unsigned short* __restrict__ out, int K) {
  size_t idx = (size_t)blockIdx.x * blockDim.x + threadIdx.x;
  int tile   = (int)(idx >> 9);
  int within = (int)(idx & 511);
  int lane = within >> 4, e = within & 15;
  int v = e >> 1, pos = e & 1;
  int sel = lane >> 4, mi = lane & 15;
  int grp = v >> 2;
  int ki = sel * 16 + v * 2 + pos;
  int tpr = K >> 5;
  int mt = tile / tpr, kt = tile % tpr;
  int m = mt * 16 + mi, k = kt * 32 + ki;
  out[idx] = f2bf(in[(size_t)m * K + k]);
}

__global__ void pack_b_kernel(const float* __restrict__ in,
                              unsigned short* __restrict__ out, int K, int N) {
  size_t idx = (size_t)blockIdx.x * blockDim.x + threadIdx.x;
  int tile   = (int)(idx >> 9);
  int within = (int)(idx & 511);
  int lane = within >> 4, e = within & 15;
  int v = e >> 1, pos = e & 1;
  int nl = lane & 15, hf = lane >> 4;
  int ki = hf * 16 + v * 2 + pos;
  int tpc = K >> 5;
  int nt = tile / tpc, kt = tile % tpc;
  int k = kt * 32 + ki, n = nt * 16 + nl;
  out[idx] = f2bf(in[(size_t)k * N + n]);
}

// ---------------- stage 2: QKV projection GEMM (bf16 WMMA) ----------------
// One wave computes a 64x64 tile of Y = X*W + b (4x4 grid of 16x16 WMMA
// accumulators): each K-step loads 8 fragments and issues 16 WMMAs, a 4x
// better WMMA-per-byte ratio than a 16x16-per-wave scheme. Epilogue scatters
// into per-head attention-ready fragment layouts:
//   which==0 (Q): A-frag of [2048 x 64]   per (b, head)
//   which==1 (K): B-frag of K^T [64 x 2048] per (b, head)
//   which==2 (V): B-frag of V  [2048 x 64]  per (b, head)
__global__ __launch_bounds__(256) void qkv_gemm_kernel(
    const unsigned short* __restrict__ Xfrom, const unsigned short* __restrict__ Xto,
    const unsigned short* __restrict__ Wfrag,
    const float* __restrict__ bq, const float* __restrict__ bk, const float* __restrict__ bv,
    unsigned short* __restrict__ Qf, unsigned short* __restrict__ KTf,
    unsigned short* __restrict__ Vf) {
  const int tilesM4 = ROWS / 64;               // 64   (64-row macro tiles)
  const int tilesN4 = D_MODEL / 64;            // 16   (64-col macro tiles)
  const int tilesPerMat = tilesM4 * tilesN4;   // 1024
  int lane = threadIdx.x & 31;
  int wid  = blockIdx.x * (blockDim.x >> 5) + (threadIdx.x >> 5);
  int which = wid / tilesPerMat;               // 0=Q 1=K 2=V (grid sized exactly)
  int t     = wid % tilesPerMat;
  int mt4 = t / tilesN4, nt4 = t % tilesN4;

  const unsigned short* X = (which == 0) ? Xfrom : Xto;
  const unsigned short* W = Wfrag + (size_t)which * D_MODEL * D_MODEL;
  const float* bias = (which == 0) ? bq : ((which == 1) ? bk : bv);

  v8f acc[4][4];
#pragma unroll
  for (int i = 0; i < 4; ++i)
#pragma unroll
    for (int j = 0; j < 4; ++j)
#pragma unroll
      for (int e = 0; e < 8; ++e) acc[i][j][e] = 0.0f;

  const int ktiles = D_MODEL / 32;             // 32
  // A row-tile bases: rows (mt4*4+i)*16, tile index = (mt)*ktiles + kt
  // B col-tile bases: cols (nt4*4+j)*16, tile index = (nt)*ktiles + kt
  const unsigned short* Ab[4];
  const unsigned short* Bb[4];
#pragma unroll
  for (int i = 0; i < 4; ++i) Ab[i] = X + (size_t)((mt4 * 4 + i) * ktiles) * 512;
#pragma unroll
  for (int j = 0; j < 4; ++j) Bb[j] = W + (size_t)((nt4 * 4 + j) * ktiles) * 512;

  for (int kt = 0; kt < ktiles; ++kt) {
    if (kt + 1 < ktiles) {                     // hint next tiles into cache
#pragma unroll
      for (int i = 0; i < 4; ++i) __builtin_prefetch(Ab[i] + (size_t)(kt + 1) * 512, 0, 3);
#pragma unroll
      for (int j = 0; j < 4; ++j) __builtin_prefetch(Bb[j] + (size_t)(kt + 1) * 512, 0, 3);
    }
    v16bf a[4], b[4];
#pragma unroll
    for (int i = 0; i < 4; ++i) a[i] = load_frag(Ab[i] + (size_t)kt * 512, lane);
#pragma unroll
    for (int j = 0; j < 4; ++j) b[j] = load_frag(Bb[j] + (size_t)kt * 512, lane);
#pragma unroll
    for (int i = 0; i < 4; ++i)
#pragma unroll
      for (int j = 0; j < 4; ++j) acc[i][j] = wmma_bf16(a[i], b[j], acc[i][j]);
  }

  int half = lane >> 4;
#pragma unroll
  for (int j = 0; j < 4; ++j) {
    int ncol = (nt4 * 4 + j) * 16 + (lane & 15);
    float bcol = bias[ncol];
    int head = ncol >> 6;                      // which head this column feeds
    int hd   = ncol & 63;                      // position within head
#pragma unroll
    for (int i = 0; i < 4; ++i) {
#pragma unroll
      for (int r = 0; r < 8; ++r) {
        int mrow = (mt4 * 4 + i) * 16 + r + 8 * half;
        int bb = mrow >> 11;                   // batch
        int f  = mrow & (SEQ - 1);             // sequence position
        unsigned short val = f2bf(acc[i][j][r] + bcol);
        size_t headbase = (size_t)(bb * N_HEADS + head) * (SEQ * HEAD_DIM);
        if (which == 0) {
          Qf[headbase + offA(f, hd, HEAD_DIM)] = val;    // Q[f, hd] as A-frag
        } else if (which == 1) {
          KTf[headbase + offB(hd, f, HEAD_DIM)] = val;   // K^T[hd, f] as B-frag
        } else {
          Vf[headbase + offB(f, hd, SEQ)] = val;         // V[f, hd] as B-frag
        }
      }
    }
  }
}

// ---------------- stage 3: flash attention (bf16 WMMA, online softmax) ------
// One wave per (batch, head, 16-query tile); 4 waves per block.
__global__ __launch_bounds__(128) void attention_kernel(
    const unsigned short* __restrict__ Qf, const unsigned short* __restrict__ KTf,
    const unsigned short* __restrict__ Vf, const int* __restrict__ mask,
    float* __restrict__ out) {
  __shared__ __align__(16) unsigned short ldsP[4][16 * 32];  // per-wave P staging

  int lane = threadIdx.x & 31;
  int wib  = threadIdx.x >> 5;
  int wid  = blockIdx.x * 4 + wib;           // 0..4095
  int ft = wid & 127;                        // query tile (16 rows)
  int hh = (wid >> 7) & (N_HEADS - 1);
  int bb = wid >> 11;

  size_t headbase = (size_t)(bb * N_HEADS + hh) * (SEQ * HEAD_DIM);
  const unsigned short* Qh = Qf + headbase;
  const unsigned short* Kh = KTf + headbase;
  const unsigned short* Vh = Vf + headbase;
  const int* mb = mask + (size_t)bb * SEQ * SEQ;

  int half = lane >> 4, lncol = lane & 15;

  // Q fragments for this query tile (K dim = HEAD_DIM = 64 -> 2 A-frags), kept live.
  v16bf qa0 = load_frag(Qh + (size_t)(ft * 2 + 0) * 512, lane);
  v16bf qa1 = load_frag(Qh + (size_t)(ft * 2 + 1) * 512, lane);

  v8f o0, o1, o2, o3;
  float Mr[8], Lr[8];
#pragma unroll
  for (int i = 0; i < 8; ++i) {
    o0[i] = 0.f; o1[i] = 0.f; o2[i] = 0.f; o3[i] = 0.f;
    Mr[i] = -3.0e38f; Lr[i] = 0.f;
  }

  for (int t = 0; t < SEQ; t += 32) {
    int tt0 = t >> 4, tt1 = tt0 + 1;
    // ---- S = Q K^T for 32 keys (two 16x16 score tiles, 4 WMMAs) ----
    v8f s0, s1;
#pragma unroll
    for (int i = 0; i < 8; ++i) { s0[i] = 0.f; s1[i] = 0.f; }
    s0 = wmma_bf16(qa0, load_frag(Kh + (size_t)(tt0 * 2 + 0) * 512, lane), s0);
    s0 = wmma_bf16(qa1, load_frag(Kh + (size_t)(tt0 * 2 + 1) * 512, lane), s0);
    s1 = wmma_bf16(qa0, load_frag(Kh + (size_t)(tt1 * 2 + 0) * 512, lane), s1);
    s1 = wmma_bf16(qa1, load_frag(Kh + (size_t)(tt1 * 2 + 1) * 512, lane), s1);

    // ---- scale + additive mask, online softmax, stage P to LDS ----
    int c0 = t + lncol, c1 = c0 + 16;
#pragma unroll
    for (int r = 0; r < 8; ++r) {
      int f = ft * 16 + r + 8 * half;
      const int* mr = mb + (size_t)f * SEQ;
      float a0 = (1.0f - (float)mr[c0]) * -10000.0f;
      float a1 = (1.0f - (float)mr[c1]) * -10000.0f;
      float v0 = s0[r] * 0.125f + a0;        // 1/sqrt(64)
      float v1 = s1[r] * 0.125f + a1;

      float mx   = redmax16(fmaxf(v0, v1)); // row max across 16 lanes (key dim)
      float Mnew = fmaxf(Mr[r], mx);
      float alpha = __expf(Mr[r] - Mnew);
      float p0 = __expf(v0 - Mnew);
      float p1 = __expf(v1 - Mnew);
      float rs = redsum16(p0 + p1);
      Lr[r] = Lr[r] * alpha + rs;
      Mr[r] = Mnew;
      o0[r] *= alpha; o1[r] *= alpha; o2[r] *= alpha; o3[r] *= alpha;

      int rowl = r + 8 * half;               // transpose C-layout -> row-major P
      ldsP[wib][rowl * 32 + lncol]      = f2bf(p0);
      ldsP[wib][rowl * 32 + 16 + lncol] = f2bf(p1);
    }
    __asm__ volatile("" ::: "memory");       // keep DS store->load program order
    // wave-private LDS; per-wave DS ops are in-order, reload P as an A-frag
    FragU pu;
    pu.q[0] = *reinterpret_cast<const uint4*>(&ldsP[wib][lncol * 32 + half * 16]);
    pu.q[1] = *reinterpret_cast<const uint4*>(&ldsP[wib][lncol * 32 + half * 16 + 8]);
    v16bf pa = pu.v;

    // ---- O += P V (K dim = 32 keys, 4 head-dim tiles) ----
    int vkt = t >> 5;
    o0 = wmma_bf16(pa, load_frag(Vh + (size_t)(0 * (SEQ / 32) + vkt) * 512, lane), o0);
    o1 = wmma_bf16(pa, load_frag(Vh + (size_t)(1 * (SEQ / 32) + vkt) * 512, lane), o1);
    o2 = wmma_bf16(pa, load_frag(Vh + (size_t)(2 * (SEQ / 32) + vkt) * 512, lane), o2);
    o3 = wmma_bf16(pa, load_frag(Vh + (size_t)(3 * (SEQ / 32) + vkt) * 512, lane), o3);
  }

  // ---- epilogue: normalize by l, write fp32 [B, F, N*H] ----
#pragma unroll
  for (int r = 0; r < 8; ++r) {
    int f = ft * 16 + r + 8 * half;
    float inv = 1.0f / Lr[r];
    size_t rowbase = ((size_t)bb * SEQ + f) * D_MODEL + hh * HEAD_DIM + lncol;
    out[rowbase + 0]  = o0[r] * inv;
    out[rowbase + 16] = o1[r] * inv;
    out[rowbase + 32] = o2[r] * inv;
    out[rowbase + 48] = o3[r] * inv;
  }
}

// ---------------- host launcher ----------------
extern "C" void kernel_launch(void* const* d_in, const int* in_sizes, int n_in,
                              void* d_out, int out_size, void* d_ws, size_t ws_size,
                              hipStream_t stream) {
  const float* from = (const float*)d_in[0];
  const float* to   = (const float*)d_in[1];
  const int*   mask = (const int*)d_in[2];
  const float* Wq = (const float*)d_in[3];
  const float* bq = (const float*)d_in[4];
  const float* Wk = (const float*)d_in[5];
  const float* bk = (const float*)d_in[6];
  const float* Wv = (const float*)d_in[7];
  const float* bv = (const float*)d_in[8];
  float* out = (float*)d_out;

  // workspace partition (bf16 fragment buffers)
  char* ws = (char*)d_ws;
  unsigned short* Xfrom = (unsigned short*)(ws + (size_t)0);            // 8 MiB
  unsigned short* Xto   = (unsigned short*)(ws + ((size_t)8  << 20));   // 8 MiB
  unsigned short* Wf    = (unsigned short*)(ws + ((size_t)16 << 20));   // 6 MiB (3x2)
  unsigned short* Qf    = (unsigned short*)(ws + ((size_t)22 << 20));   // 8 MiB
  unsigned short* KTf   = (unsigned short*)(ws + ((size_t)30 << 20));   // 8 MiB
  unsigned short* Vf    = (unsigned short*)(ws + ((size_t)38 << 20));   // 8 MiB

  const size_t xElems = (size_t)ROWS * D_MODEL;      // 4 Mi
  const size_t wElems = (size_t)D_MODEL * D_MODEL;   // 1 Mi

  // stage 1: pack activations (A-frag) and weights (B-frag) as bf16
  pack_a_kernel<<<(unsigned)(xElems / 256), 256, 0, stream>>>(from, Xfrom, D_MODEL);
  pack_a_kernel<<<(unsigned)(xElems / 256), 256, 0, stream>>>(to, Xto, D_MODEL);
  pack_b_kernel<<<(unsigned)(wElems / 256), 256, 0, stream>>>(Wq, Wf + 0 * wElems, D_MODEL, D_MODEL);
  pack_b_kernel<<<(unsigned)(wElems / 256), 256, 0, stream>>>(Wk, Wf + 1 * wElems, D_MODEL, D_MODEL);
  pack_b_kernel<<<(unsigned)(wElems / 256), 256, 0, stream>>>(Wv, Wf + 2 * wElems, D_MODEL, D_MODEL);

  // stage 2: QKV projections; 3 * 64 * 16 = 3072 macro-tiles, 8 waves/block
  qkv_gemm_kernel<<<3072 / 8, 256, 0, stream>>>(Xfrom, Xto, Wf, bq, bk, bv, Qf, KTf, Vf);

  // stage 3: attention; 2*16*128 = 4096 waves, 4 waves/block
  attention_kernel<<<4096 / 4, 128, 0, stream>>>(Qf, KTf, Vf, mask, out);
}
